// GAT_75703093559414
// MI455X (gfx1250) — matrix operand order repaired and checked by reading, not compiled, not run on hardware
//
#include <hip/hip_runtime.h>
#include <hip/hip_fp16.h>
#include <math.h>

typedef __attribute__((ext_vector_type(16))) _Float16 v16h;
typedef __attribute__((ext_vector_type(8)))  float    v8f;

#define IN_CH 128
#define C1 256           // HEADS*HID
#define HEADS 8
#define HID 32
#define NEG_SLOPE 0.2f

// ---------- helpers ----------
__device__ __forceinline__ float lrelu(float x) { return x > 0.f ? x : NEG_SLOPE * x; }
// monotone float<->int key so atomicMax(int*) orders floats correctly
__device__ __forceinline__ int   fkey(float x)  { int b = __float_as_int(x); return b >= 0 ? b : (int)(b ^ 0x7fffffff); }
__device__ __forceinline__ float funkey(int k)  { return __int_as_float(k >= 0 ? k : (int)(k ^ 0x7fffffff)); }

// ---------- init ----------
__global__ void fill_f32(float* p, int n, float v) {
    for (int i = blockIdx.x * blockDim.x + threadIdx.x; i < n; i += gridDim.x * blockDim.x) p[i] = v;
}
__global__ void fill_key(int* p, int n) {
    int k = fkey(-3.402823e38f);
    for (int i = blockIdx.x * blockDim.x + threadIdx.x; i < n; i += gridDim.x * blockDim.x) p[i] = k;
}

// ---------- W1 transpose + f32->f16 : W1T[col*128 + k] = W1[k*256 + col] ----------
__global__ void w1_transpose(const float* __restrict__ W1, _Float16* __restrict__ W1T) {
    int tid = blockIdx.x * blockDim.x + threadIdx.x;
    if (tid >= C1 * IN_CH) return;
    int col = tid >> 7, k = tid & 127;
    W1T[tid] = (_Float16)W1[k * C1 + col];
}

// ---------- GEMM1: h1[N,256] = x[N,128] @ W1[128,256] via v_wmma_f32_16x16x32_f16 ----------
// 1 block = 16 rows x 256 cols.  8 waves; wave w owns cols [32w,32w+32) as two 16x16 tiles.
__global__ void gemm1_wmma(const float* __restrict__ x, const _Float16* __restrict__ W1T,
                           float* __restrict__ h1, int N) {
    int lane = threadIdx.x & 31;
    int wave = threadIdx.x >> 5;
    int r0   = blockIdx.x * 16;
    int m    = lane & 15;
    int g    = lane >> 4;            // half-wave group
    int row  = r0 + m; if (row >= N) row = N - 1;
    int cb   = wave * 32;

    v8f c0 = {}; v8f c1 = {};
    const float* xrow = x + (size_t)row * IN_CH;

    for (int kk = 0; kk < IN_CH; kk += 32) {
        // A fragment (16-bit A 16x32 layout): lane holds k in [16g,16g+16) (contiguous, matches B)
        float4 f0 = *(const float4*)(xrow + kk + 16 * g);
        float4 f1 = *(const float4*)(xrow + kk + 16 * g + 4);
        float4 f2 = *(const float4*)(xrow + kk + 16 * g + 8);
        float4 f3 = *(const float4*)(xrow + kk + 16 * g + 12);
        v16h a;
        a[0]=(_Float16)f0.x;  a[1]=(_Float16)f0.y;  a[2]=(_Float16)f0.z;  a[3]=(_Float16)f0.w;
        a[4]=(_Float16)f1.x;  a[5]=(_Float16)f1.y;  a[6]=(_Float16)f1.z;  a[7]=(_Float16)f1.w;
        a[8]=(_Float16)f2.x;  a[9]=(_Float16)f2.y;  a[10]=(_Float16)f2.z; a[11]=(_Float16)f2.w;
        a[12]=(_Float16)f3.x; a[13]=(_Float16)f3.y; a[14]=(_Float16)f3.z; a[15]=(_Float16)f3.w;

        // B fragment (16-bit B 32x16): lane holds col = cb+m, k = 16g + j  -> contiguous in W1T
        int colA = cb + m;
        v16h b0 = *(const v16h*)(W1T + (size_t)colA        * IN_CH + kk + 16 * g);
        v16h b1 = *(const v16h*)(W1T + (size_t)(colA + 16) * IN_CH + kk + 16 * g);

        c0 = __builtin_amdgcn_wmma_f32_16x16x32_f16(false, a, false, b0, (short)0, c0, false, false);
        c1 = __builtin_amdgcn_wmma_f32_16x16x32_f16(false, a, false, b1, (short)0, c1, false, false);
    }

    // C/D layout: VGPR r -> row = r + 8g, col = lane&15
    for (int r = 0; r < 8; r++) {
        int orow = r0 + r + 8 * g;
        if (orow < N) {
            h1[(size_t)orow * C1 + cb + m]      = c0[r];
            h1[(size_t)orow * C1 + cb + 16 + m] = c1[r];
        }
    }
}

// ---------- attention coefficients: a_src/a_dst [N,8] ----------
__global__ void att1_kernel(const float* __restrict__ h1, const float* __restrict__ atts,
                            const float* __restrict__ attd, float* __restrict__ as,
                            float* __restrict__ ad, int N) {
    int tid = blockIdx.x * blockDim.x + threadIdx.x;
    if (tid >= N * HEADS) return;
    int n = tid >> 3, h = tid & 7;
    const float4* hp = (const float4*)(h1 + (size_t)n * C1 + h * HID);
    const float4* sp = (const float4*)(atts + h * HID);
    const float4* dp = (const float4*)(attd + h * HID);
    float s = 0.f, d = 0.f;
    for (int q = 0; q < 8; q++) {
        float4 a = hp[q], b = sp[q], c = dp[q];
        s += a.x * b.x + a.y * b.y + a.z * b.z + a.w * b.w;
        d += a.x * c.x + a.y * c.y + a.z * c.z + a.w * c.w;
    }
    as[tid] = s; ad[tid] = d;
}

// ---------- layer-1 edge passes (per edge, per head) ----------
__device__ __forceinline__ void edge_sd(const long long* ei, int e, int E, int& s, int& d) {
    if (e < E) { s = (int)ei[e]; d = (int)ei[E + e]; } else { s = d = e - E; }
}

__global__ void edge_max1(const long long* __restrict__ ei, const float* __restrict__ as,
                          const float* __restrict__ ad, int* __restrict__ mkey, int E, int Etot) {
    int tid = blockIdx.x * blockDim.x + threadIdx.x;
    if (tid >= Etot * HEADS) return;
    int e = tid >> 3, h = tid & 7, s, d;
    edge_sd(ei, e, E, s, d);
    float v = lrelu(as[s * HEADS + h] + ad[d * HEADS + h]);
    atomicMax(&mkey[d * HEADS + h], fkey(v));
}

__global__ void edge_den1(const long long* __restrict__ ei, const float* __restrict__ as,
                          const float* __restrict__ ad, const int* __restrict__ mkey,
                          float* __restrict__ denom, int E, int Etot) {
    int tid = blockIdx.x * blockDim.x + threadIdx.x;
    if (tid >= Etot * HEADS) return;
    int e = tid >> 3, h = tid & 7, s, d;
    edge_sd(ei, e, E, s, d);
    float v = lrelu(as[s * HEADS + h] + ad[d * HEADS + h]);
    atomicAdd(&denom[d * HEADS + h], expf(v - funkey(mkey[d * HEADS + h])));
}

__global__ void edge_agg1(const long long* __restrict__ ei, const float* __restrict__ as,
                          const float* __restrict__ ad, const int* __restrict__ mkey,
                          const float* __restrict__ denom, const float* __restrict__ h1,
                          float* __restrict__ out1, int E, int Etot) {
    int tid = blockIdx.x * blockDim.x + threadIdx.x;
    if (tid >= Etot * HEADS) return;
    int e = tid >> 3, h = tid & 7, s, d;
    edge_sd(ei, e, E, s, d);
    float v = lrelu(as[s * HEADS + h] + ad[d * HEADS + h]);
    float alpha = expf(v - funkey(mkey[d * HEADS + h])) / denom[d * HEADS + h];
    const float4* sp = (const float4*)(h1 + (size_t)s * C1 + h * HID);
    float* dp = out1 + (size_t)d * C1 + h * HID;
    for (int q = 0; q < 8; q++) {
        float4 f = sp[q];
        atomicAdd(dp + 4 * q + 0, alpha * f.x);
        atomicAdd(dp + 4 * q + 1, alpha * f.y);
        atomicAdd(dp + 4 * q + 2, alpha * f.z);
        atomicAdd(dp + 4 * q + 3, alpha * f.w);
    }
}

// ---------- per-node: t = elu(out1 + b1) @ W2 ; layer-2 attention scalars ----------
__global__ void node_l2(const float* __restrict__ out1, const float* __restrict__ b1,
                        const float* __restrict__ W2, const float* __restrict__ as2c,
                        const float* __restrict__ ad2c, float* __restrict__ t,
                        float* __restrict__ a2s, float* __restrict__ a2d, int N) {
    int lane = threadIdx.x & 31;
    int node = blockIdx.x * 8 + (threadIdx.x >> 5);
    if (node >= N) return;
    const float* p = out1 + (size_t)node * C1;
    float acc = 0.f;
    for (int q = 0; q < 8; q++) {
        int j = lane + 32 * q;
        float v = p[j] + b1[j];
        v = v > 0.f ? v : (expf(v) - 1.f);   // elu
        acc += v * W2[j];
    }
    for (int off = 16; off > 0; off >>= 1) acc += __shfl_xor(acc, off, 32);
    if (lane == 0) { t[node] = acc; a2s[node] = acc * as2c[0]; a2d[node] = acc * ad2c[0]; }
}

// ---------- layer-2 edge passes (1 head) ----------
__global__ void edge_max2(const long long* __restrict__ ei, const float* __restrict__ a2s,
                          const float* __restrict__ a2d, int* __restrict__ mkey, int E, int Etot) {
    int e = blockIdx.x * blockDim.x + threadIdx.x;
    if (e >= Etot) return;
    int s, d; edge_sd(ei, e, E, s, d);
    atomicMax(&mkey[d], fkey(lrelu(a2s[s] + a2d[d])));
}
__global__ void edge_den2(const long long* __restrict__ ei, const float* __restrict__ a2s,
                          const float* __restrict__ a2d, const int* __restrict__ mkey,
                          float* __restrict__ denom, int E, int Etot) {
    int e = blockIdx.x * blockDim.x + threadIdx.x;
    if (e >= Etot) return;
    int s, d; edge_sd(ei, e, E, s, d);
    float v = lrelu(a2s[s] + a2d[d]);
    atomicAdd(&denom[d], expf(v - funkey(mkey[d])));
}
__global__ void edge_agg2(const long long* __restrict__ ei, const float* __restrict__ a2s,
                          const float* __restrict__ a2d, const int* __restrict__ mkey,
                          const float* __restrict__ denom, const float* __restrict__ t,
                          float* __restrict__ out2, int E, int Etot) {
    int e = blockIdx.x * blockDim.x + threadIdx.x;
    if (e >= Etot) return;
    int s, d; edge_sd(ei, e, E, s, d);
    float v = lrelu(a2s[s] + a2d[d]);
    float alpha = expf(v - funkey(mkey[d])) / denom[d];
    atomicAdd(&out2[d], alpha * t[s]);
}

__global__ void finalize(const float* __restrict__ out2, const float* __restrict__ b2,
                         float* __restrict__ out, int N) {
    int n = blockIdx.x * blockDim.x + threadIdx.x;
    if (n < N) out[n] = out2[n] + b2[0];
}

// ---------- launch ----------
extern "C" void kernel_launch(void* const* d_in, const int* in_sizes, int n_in,
                              void* d_out, int out_size, void* d_ws, size_t ws_size,
                              hipStream_t stream) {
    const float*     x    = (const float*)d_in[0];
    const long long* ei   = (const long long*)d_in[1];
    const float*     W1   = (const float*)d_in[2];
    const float*     atts = (const float*)d_in[3];
    const float*     attd = (const float*)d_in[4];
    const float*     b1   = (const float*)d_in[5];
    const float*     W2   = (const float*)d_in[6];
    const float*     as2  = (const float*)d_in[7];
    const float*     ad2  = (const float*)d_in[8];
    const float*     b2   = (const float*)d_in[9];
    float* out = (float*)d_out;

    const int N = in_sizes[0] / IN_CH;      // 50000
    const int E = in_sizes[1] / 2;          // 800000
    const int Etot = E + N;                 // self-loops appended

    // workspace carve-out (256B aligned)
    size_t off = 0;
    auto carve = [&](size_t bytes) {
        void* p = (char*)d_ws + off;
        off += (bytes + 255) & ~(size_t)255;
        return p;
    };
    float*     h1     = (float*)carve((size_t)N * C1 * 4);
    float*     out1   = (float*)carve((size_t)N * C1 * 4);
    _Float16*  W1T    = (_Float16*)carve((size_t)C1 * IN_CH * 2);
    float*     as1    = (float*)carve((size_t)N * HEADS * 4);
    float*     ad1    = (float*)carve((size_t)N * HEADS * 4);
    int*       mkey1  = (int*)carve((size_t)N * HEADS * 4);
    float*     den1   = (float*)carve((size_t)N * HEADS * 4);
    float*     t      = (float*)carve((size_t)N * 4);
    float*     a2sv   = (float*)carve((size_t)N * 4);
    float*     a2dv   = (float*)carve((size_t)N * 4);
    int*       mkey2  = (int*)carve((size_t)N * 4);
    float*     den2   = (float*)carve((size_t)N * 4);
    float*     out2   = (float*)carve((size_t)N * 4);
    (void)ws_size; (void)n_in; (void)out_size;

    const int TB = 256;
    // init accumulators (every call, so replay is self-consistent)
    fill_f32<<<2048, TB, 0, stream>>>(out1, N * C1, 0.f);
    fill_f32<<<256,  TB, 0, stream>>>(den1, N * HEADS, 0.f);
    fill_key<<<256,  TB, 0, stream>>>(mkey1, N * HEADS);
    fill_f32<<<64,   TB, 0, stream>>>(den2, N, 0.f);
    fill_f32<<<64,   TB, 0, stream>>>(out2, N, 0.f);
    fill_key<<<64,   TB, 0, stream>>>(mkey2, N);

    // layer 1
    w1_transpose<<<(C1 * IN_CH + TB - 1) / TB, TB, 0, stream>>>(W1, W1T);
    gemm1_wmma<<<(N + 15) / 16, TB, 0, stream>>>(x, W1T, h1, N);
    att1_kernel<<<(N * HEADS + TB - 1) / TB, TB, 0, stream>>>(h1, atts, attd, as1, ad1, N);

    int ew = Etot * HEADS;
    edge_max1<<<(ew + TB - 1) / TB, TB, 0, stream>>>(ei, as1, ad1, mkey1, E, Etot);
    edge_den1<<<(ew + TB - 1) / TB, TB, 0, stream>>>(ei, as1, ad1, mkey1, den1, E, Etot);
    edge_agg1<<<(ew + TB - 1) / TB, TB, 0, stream>>>(ei, as1, ad1, mkey1, den1, h1, out1, E, Etot);

    // layer 2
    node_l2<<<(N + 7) / 8, TB, 0, stream>>>(out1, b1, W2, as2, ad2, t, a2sv, a2dv, N);
    edge_max2<<<(Etot + TB - 1) / TB, TB, 0, stream>>>(ei, a2sv, a2dv, mkey2, E, Etot);
    edge_den2<<<(Etot + TB - 1) / TB, TB, 0, stream>>>(ei, a2sv, a2dv, mkey2, den2, E, Etot);
    edge_agg2<<<(Etot + TB - 1) / TB, TB, 0, stream>>>(ei, a2sv, a2dv, mkey2, den2, t, out2, E, Etot);
    finalize<<<(N + TB - 1) / TB, TB, 0, stream>>>(out2, b2, out, N);
}
